// GAT2_48524540510804
// MI455X (gfx1250) — matrix operand order repaired and checked by reading, not compiled, not run on hardware
//
#include <hip/hip_runtime.h>
#include <hip/hip_bf16.h>
#include <math.h>

// ---------------------------------------------------------------------------
// 2-layer GAT forward for MI455X (gfx1250, wave32).
//  * Dense projections: bf16 WMMA (v_wmma_f32_16x16x32_bf16), f32 accumulate.
//  * Edge softmax/aggregation: f32 atomics resolving in the 192MB L2.
// ---------------------------------------------------------------------------

#define GAT_SLOPE 0.25f

typedef __bf16 bf16_t;
typedef __attribute__((ext_vector_type(16))) __bf16 v16bf;
typedef __attribute__((ext_vector_type(8)))  float  v8f;

__device__ __forceinline__ float atomicMaxF(float* addr, float val) {
  // Standard sign-split trick: correct for mixed-sign values.
  if (val >= 0.0f)
    return __int_as_float(atomicMax((int*)addr, __float_as_int(val)));
  else
    return __uint_as_float(atomicMin((unsigned int*)addr, __float_as_uint(val)));
}

__device__ __forceinline__ float leaky(float x) { return x > 0.0f ? x : GAT_SLOPE * x; }

// ---------------------------------------------------------------------------
// C[M,64] = A[M,K] * B[K,64]   (all f32 row-major in memory; bf16 WMMA inside)
// Requires M % 16 == 0 (N = 50000 = 3125*16) and K % 32 == 0 (512, 64).
// Block = 128 threads = 4 waves; block covers 16 rows x 64 cols,
// wave w owns the 16-column tile starting at 16*w.
// ---------------------------------------------------------------------------
__global__ void __launch_bounds__(128)
wmma_gemm64_kernel(const float* __restrict__ A, const float* __restrict__ B,
                   float* __restrict__ C, int M, int K) {
  const int lane    = threadIdx.x & 31;
  const int wave    = threadIdx.x >> 5;
  const int rowBase = blockIdx.x * 16;
  const int colBase = wave * 16;
  if (rowBase >= M) return;

  const int m    = lane & 15;   // A row within tile / B,C column within tile
  const int half = lane >> 4;   // K-half selector per ISA layout

  const float* arow = A + (size_t)(rowBase + m) * K;
  v8f acc = {};

  for (int k0 = 0; k0 < K; k0 += 32) {
    // A fragment: 16x32 bf16. Lane holds M=lane&15; VGPR0..3 = K half*8+0..7,
    // VGPR4..7 = K 16+half*8+0..7 (2 bf16 per VGPR).
    v16bf a, b;
    const float* ap = arow + k0 + half * 8;
#pragma unroll
    for (int i = 0; i < 8; ++i) {
      a[i]     = (bf16_t)ap[i];
      a[i + 8] = (bf16_t)ap[i + 16];
    }
    // B fragment: 32x16 bf16. n = lane&15; same K order as A:
    // elements 0..7 = K half*8+0..7, elements 8..15 = K 16+half*8+0..7.
    const float* bp = B + (size_t)(k0 + half * 8) * 64 + colBase + m;
#pragma unroll
    for (int i = 0; i < 16; ++i) b[i] = (bf16_t)bp[(size_t)(i + (i & 8)) * 64];

    acc = __builtin_amdgcn_wmma_f32_16x16x32_bf16(false, a, false, b,
                                                  (short)0, acc, false, false);
  }

  // D layout: VGPR v -> row (v + 8*half), column lane&15.
  const int n = colBase + m;
#pragma unroll
  for (int v = 0; v < 8; ++v)
    C[(size_t)(rowBase + v + 8 * half) * 64 + n] = acc[v];
}

// ---------------------------------------------------------------------------
// Per-node attention logits: als[n,h] = <H[n,h,:], a_src[h,:]>, same for dst.
// ---------------------------------------------------------------------------
__global__ void attn_logit_kernel(const float* __restrict__ H,
                                  const float* __restrict__ a_src,
                                  const float* __restrict__ a_dst,
                                  float* __restrict__ als, float* __restrict__ ald,
                                  int N, int heads, int cout) {
  int idx = blockIdx.x * blockDim.x + threadIdx.x;
  if (idx >= N * heads) return;
  int n = idx / heads, h = idx - n * heads;
  const float* hp = H + (size_t)n * heads * cout + h * cout;
  const float* sp = a_src + h * cout;
  const float* dp = a_dst + h * cout;
  float s = 0.0f, d = 0.0f;
  for (int c = 0; c < cout; ++c) { float v = hp[c]; s += v * sp[c]; d += v * dp[c]; }
  als[idx] = s;
  ald[idx] = d;
}

__global__ void fill_kernel(float* __restrict__ p, float v, long long n) {
  long long i = (long long)blockIdx.x * blockDim.x + threadIdx.x;
  if (i < n) p[i] = v;
}

// segment max of leaky(als[src]+ald[dst]) over dst
__global__ void seg_max_kernel(const int* __restrict__ src, const int* __restrict__ dst,
                               long long E, int H,
                               const float* __restrict__ als, const float* __restrict__ ald,
                               float* __restrict__ m) {
  long long idx = (long long)blockIdx.x * blockDim.x + threadIdx.x;
  if (idx >= E * H) return;
  int e = (int)(idx / H), h = (int)(idx - (long long)e * H);
  int s = src[e], d = dst[e];
  float lg = leaky(als[(size_t)s * H + h] + ald[(size_t)d * H + h]);
  atomicMaxF(&m[(size_t)d * H + h], lg);
}

// segment sum of exp(logit - m[dst])
__global__ void seg_sum_kernel(const int* __restrict__ src, const int* __restrict__ dst,
                               long long E, int H,
                               const float* __restrict__ als, const float* __restrict__ ald,
                               const float* __restrict__ m, float* __restrict__ z) {
  long long idx = (long long)blockIdx.x * blockDim.x + threadIdx.x;
  if (idx >= E * H) return;
  int e = (int)(idx / H), h = (int)(idx - (long long)e * H);
  int s = src[e], d = dst[e];
  float lg = leaky(als[(size_t)s * H + h] + ald[(size_t)d * H + h]);
  atomicAdd(&z[(size_t)d * H + h], expf(lg - m[(size_t)d * H + h]));
}

// out[dst, c] += alpha(e, c/cout) * Hf[src, c]   (64 channels both layers)
__global__ void aggregate_kernel(const int* __restrict__ src, const int* __restrict__ dst,
                                 long long E, int H, int cout,
                                 const float* __restrict__ als, const float* __restrict__ ald,
                                 const float* __restrict__ m, const float* __restrict__ z,
                                 const float* __restrict__ Hf, float* __restrict__ out) {
  long long idx = (long long)blockIdx.x * blockDim.x + threadIdx.x;
  if (idx >= E * 64) return;
  int e = (int)(idx >> 6);
  int c = (int)(idx & 63);
  int h = c / cout;
  int s = src[e], d = dst[e];
  float lg    = leaky(als[(size_t)s * H + h] + ald[(size_t)d * H + h]);
  float alpha = expf(lg - m[(size_t)d * H + h]) / (z[(size_t)d * H + h] + 1e-16f);
  atomicAdd(&out[(size_t)d * 64 + c], alpha * Hf[(size_t)s * 64 + c]);
}

// mid = softmax(aggr_w)[0]*elu(agg1+b1) + softmax(aggr_w)[1]*elu(agg2+b2)
__global__ void mid_kernel(const float* __restrict__ agg1, const float* __restrict__ agg2,
                           const float* __restrict__ b1, const float* __restrict__ b2,
                           const float* __restrict__ aggr_w, float* __restrict__ mid,
                           long long total) {
  long long idx = (long long)blockIdx.x * blockDim.x + threadIdx.x;
  if (idx >= total) return;
  int c = (int)(idx & 63);
  float w0 = aggr_w[c], w1 = aggr_w[64 + c];
  float e0 = expf(w0), e1 = expf(w1);
  float inv = 1.0f / (e0 + e1);
  float x1 = agg1[idx] + b1[c];
  float x2 = agg2[idx] + b2[c];
  x1 = x1 > 0.0f ? x1 : expf(x1) - 1.0f;   // elu
  x2 = x2 > 0.0f ? x2 : expf(x2) - 1.0f;
  mid[idx] = (e0 * x1 + e1 * x2) * inv;
}

// final combine + per-row log_softmax over 64 channels (one wave32 per node)
__global__ void final_kernel(const float* __restrict__ aggA, const float* __restrict__ aggB,
                             const float* __restrict__ bA, const float* __restrict__ bB,
                             const float* __restrict__ aggr_w, float* __restrict__ out, int N) {
  int wpb  = blockDim.x >> 5;
  int node = blockIdx.x * wpb + (threadIdx.x >> 5);
  if (node >= N) return;
  int lane = threadIdx.x & 31;

  float v[2];
#pragma unroll
  for (int j = 0; j < 2; ++j) {
    int c = lane + 32 * j;
    float w0 = aggr_w[c], w1 = aggr_w[64 + c];
    float e0 = expf(w0), e1 = expf(w1);
    float inv = 1.0f / (e0 + e1);
    float xA = aggA[(size_t)node * 64 + c] + bA[c];
    float xB = aggB[(size_t)node * 64 + c] + bB[c];
    v[j] = (e0 * xA + e1 * xB) * inv;
  }
  float mx = fmaxf(v[0], v[1]);
#pragma unroll
  for (int off = 16; off > 0; off >>= 1) mx = fmaxf(mx, __shfl_xor(mx, off, 32));
  float se = expf(v[0] - mx) + expf(v[1] - mx);
#pragma unroll
  for (int off = 16; off > 0; off >>= 1) se += __shfl_xor(se, off, 32);
  float lse = mx + logf(se);
  out[(size_t)node * 64 + lane]      = v[0] - lse;
  out[(size_t)node * 64 + lane + 32] = v[1] - lse;
}

// ---------------------------------------------------------------------------

static inline unsigned gdim(long long total, int block) {
  return (unsigned)((total + block - 1) / block);
}

extern "C" void kernel_launch(void* const* d_in, const int* in_sizes, int n_in,
                              void* d_out, int out_size, void* d_ws, size_t ws_size,
                              hipStream_t stream) {
  const float* x     = (const float*)d_in[0];
  const int*   adj1  = (const int*)d_in[1];
  const int*   adj2  = (const int*)d_in[2];
  const float* W11   = (const float*)d_in[3];
  const float* as11  = (const float*)d_in[4];
  const float* ad11  = (const float*)d_in[5];
  const float* b11   = (const float*)d_in[6];
  const float* W21   = (const float*)d_in[7];
  const float* as21  = (const float*)d_in[8];
  const float* ad21  = (const float*)d_in[9];
  const float* b21   = (const float*)d_in[10];
  const float* W12   = (const float*)d_in[11];
  const float* as12  = (const float*)d_in[12];
  const float* ad12  = (const float*)d_in[13];
  const float* b12   = (const float*)d_in[14];
  const float* W22   = (const float*)d_in[15];
  const float* as22  = (const float*)d_in[16];
  const float* ad22  = (const float*)d_in[17];
  const float* b22   = (const float*)d_in[18];
  const float* aggr1w = (const float*)d_in[19];
  const float* aggr2w = (const float*)d_in[20];
  float* out = (float*)d_out;

  const int IN = 512;
  const long long N  = in_sizes[0] / IN;
  const long long E1 = in_sizes[1] / 2;
  const long long E2 = in_sizes[2] / 2;
  const int* src1 = adj1;             const int* dst1 = adj1 + E1;
  const int* src2 = adj2;             const int* dst2 = adj2 + E2;

  // -------- workspace partition (f32), reused across layers --------
  float* ws = (float*)d_ws;
  size_t off = 0;
  float* H1   = ws + off; off += (size_t)N * 64;   // layer1 proj A / layer2 proj A
  float* H2   = ws + off; off += (size_t)N * 64;   // layer1 proj B / layer2 proj B
  float* MID  = ws + off; off += (size_t)N * 64;
  float* agg1 = ws + off; off += (size_t)N * 64;   // reused for layer2 aggregation
  float* agg2 = ws + off; off += (size_t)N * 64;
  float* als1 = ws + off; off += (size_t)N * 8;
  float* ald1 = ws + off; off += (size_t)N * 8;
  float* als2 = ws + off; off += (size_t)N * 8;
  float* ald2 = ws + off; off += (size_t)N * 8;
  float* m1   = ws + off; off += (size_t)N * 8;
  float* z1   = ws + off; off += (size_t)N * 8;
  float* m2   = ws + off; off += (size_t)N * 8;
  float* z2   = ws + off; off += (size_t)N * 8;
  (void)ws_size; (void)n_in; (void)out_size;

  const int BLK = 256;
  const unsigned gemmGrid = (unsigned)((N + 15) / 16);

  // ================= Layer 1 =================
  wmma_gemm64_kernel<<<gemmGrid, 128, 0, stream>>>(x, W11, H1, (int)N, IN);
  wmma_gemm64_kernel<<<gemmGrid, 128, 0, stream>>>(x, W21, H2, (int)N, IN);

  attn_logit_kernel<<<gdim(N * 8, BLK), BLK, 0, stream>>>(H1, as11, ad11, als1, ald1, (int)N, 8, 8);
  attn_logit_kernel<<<gdim(N * 8, BLK), BLK, 0, stream>>>(H2, as21, ad21, als2, ald2, (int)N, 8, 8);

  fill_kernel<<<gdim(N * 8, BLK), BLK, 0, stream>>>(m1, -INFINITY, N * 8);
  fill_kernel<<<gdim(N * 8, BLK), BLK, 0, stream>>>(z1, 0.0f, N * 8);
  fill_kernel<<<gdim(N * 8, BLK), BLK, 0, stream>>>(m2, -INFINITY, N * 8);
  fill_kernel<<<gdim(N * 8, BLK), BLK, 0, stream>>>(z2, 0.0f, N * 8);
  fill_kernel<<<gdim(N * 64, BLK), BLK, 0, stream>>>(agg1, 0.0f, N * 64);
  fill_kernel<<<gdim(N * 64, BLK), BLK, 0, stream>>>(agg2, 0.0f, N * 64);

  seg_max_kernel<<<gdim(E1 * 8, BLK), BLK, 0, stream>>>(src1, dst1, E1, 8, als1, ald1, m1);
  seg_max_kernel<<<gdim(E2 * 8, BLK), BLK, 0, stream>>>(src2, dst2, E2, 8, als2, ald2, m2);
  seg_sum_kernel<<<gdim(E1 * 8, BLK), BLK, 0, stream>>>(src1, dst1, E1, 8, als1, ald1, m1, z1);
  seg_sum_kernel<<<gdim(E2 * 8, BLK), BLK, 0, stream>>>(src2, dst2, E2, 8, als2, ald2, m2, z2);

  aggregate_kernel<<<gdim(E1 * 64, BLK), BLK, 0, stream>>>(src1, dst1, E1, 8, 8,
                                                           als1, ald1, m1, z1, H1, agg1);
  aggregate_kernel<<<gdim(E2 * 64, BLK), BLK, 0, stream>>>(src2, dst2, E2, 8, 8,
                                                           als2, ald2, m2, z2, H2, agg2);

  mid_kernel<<<gdim(N * 64, BLK), BLK, 0, stream>>>(agg1, agg2, b11, b21, aggr1w, MID, N * 64);

  // ================= Layer 2 (heads=1, cout=64) =================
  wmma_gemm64_kernel<<<gemmGrid, 128, 0, stream>>>(MID, W12, H1, (int)N, 64);
  wmma_gemm64_kernel<<<gemmGrid, 128, 0, stream>>>(MID, W22, H2, (int)N, 64);

  attn_logit_kernel<<<gdim(N, BLK), BLK, 0, stream>>>(H1, as12, ad12, als1, ald1, (int)N, 1, 64);
  attn_logit_kernel<<<gdim(N, BLK), BLK, 0, stream>>>(H2, as22, ad22, als2, ald2, (int)N, 1, 64);

  fill_kernel<<<gdim(N, BLK), BLK, 0, stream>>>(m1, -INFINITY, N);
  fill_kernel<<<gdim(N, BLK), BLK, 0, stream>>>(z1, 0.0f, N);
  fill_kernel<<<gdim(N, BLK), BLK, 0, stream>>>(m2, -INFINITY, N);
  fill_kernel<<<gdim(N, BLK), BLK, 0, stream>>>(z2, 0.0f, N);
  fill_kernel<<<gdim(N * 64, BLK), BLK, 0, stream>>>(agg1, 0.0f, N * 64);
  fill_kernel<<<gdim(N * 64, BLK), BLK, 0, stream>>>(agg2, 0.0f, N * 64);

  seg_max_kernel<<<gdim(E1, BLK), BLK, 0, stream>>>(src1, dst1, E1, 1, als1, ald1, m1);
  seg_max_kernel<<<gdim(E2, BLK), BLK, 0, stream>>>(src2, dst2, E2, 1, als2, ald2, m2);
  seg_sum_kernel<<<gdim(E1, BLK), BLK, 0, stream>>>(src1, dst1, E1, 1, als1, ald1, m1, z1);
  seg_sum_kernel<<<gdim(E2, BLK), BLK, 0, stream>>>(src2, dst2, E2, 1, als2, ald2, m2, z2);

  aggregate_kernel<<<gdim(E1 * 64, BLK), BLK, 0, stream>>>(src1, dst1, E1, 1, 64,
                                                           als1, ald1, m1, z1, H1, agg1);
  aggregate_kernel<<<gdim(E2 * 64, BLK), BLK, 0, stream>>>(src2, dst2, E2, 1, 64,
                                                           als2, ald2, m2, z2, H2, agg2);

  final_kernel<<<gdim(N, BLK) * 0 + (unsigned)((N + 7) / 8), 256, 0, stream>>>(
      agg1, agg2, b12, b22, aggr2w, out, (int)N);
}
